// GCNGenerator_6648609374284
// MI455X (gfx1250) — matrix operand order repaired and checked by reading, not compiled, not run on hardware
//
#include <hip/hip_runtime.h>
#include <hip/hip_bf16.h>
#include <stdint.h>

#define N_NODES 50000
#define N_EDGES 600000
#define F_IN    128
#define F_HID   256
#define F_OUT   128
#define KPAD    50016      /* 50000 rounded up to multiple of 32 */
#define BN_EPS  1e-3f

typedef __attribute__((ext_vector_type(16))) __bf16         v16bf;
typedef __attribute__((ext_vector_type(8)))  float          v8f;
typedef __attribute__((ext_vector_type(8)))  unsigned short u16x8;
typedef __attribute__((ext_vector_type(16))) unsigned short u16x16;

union BF16Frag { u16x16 u; v16bf v; };

__device__ __forceinline__ unsigned short f2bf(float f) {
    unsigned int u = __float_as_uint(f);
    u += 0x7FFFu + ((u >> 16) & 1u);        // round-to-nearest-even
    return (unsigned short)(u >> 16);
}

// ---------------------------------------------------------------- degree
__global__ void k_deg_init(float* deg) {
    int i = blockIdx.x * 256 + threadIdx.x;
    if (i < N_NODES) deg[i] = 1.0f;         // self-loop
}

__global__ void k_deg_count(const long long* __restrict__ ei, float* __restrict__ deg) {
    int e = blockIdx.x * 256 + threadIdx.x;
    if (e < N_EDGES) atomicAdd(&deg[(int)ei[N_EDGES + e]], 1.0f);
}

__global__ void k_dinv(float* deg) {
    int i = blockIdx.x * 256 + threadIdx.x;
    if (i < N_NODES) deg[i] = rsqrtf(deg[i]);
}

// per-edge symmetric-norm coefficient: coef[e] = dinv[src]*dinv[dst]
__global__ void k_edge_coef(const long long* __restrict__ ei, const float* __restrict__ dinv,
                            float* __restrict__ coef) {
    int e = blockIdx.x * 256 + threadIdx.x;
    if (e < N_EDGES) coef[e] = dinv[(int)ei[e]] * dinv[(int)ei[N_EDGES + e]];
}

// ---------------------------------------------------------------- converts
__global__ void k_f32_to_bf16(const float* __restrict__ in, unsigned short* __restrict__ out,
                              long long n) {
    long long i = (long long)blockIdx.x * 256 + threadIdx.x;
    if (i < n) out[i] = f2bf(in[i]);
}

// W: [K][Ncols] row-major f32  ->  WT: [Ncols][K] bf16 (i.e. B column-major)
__global__ void k_transpose_bf16(const float* __restrict__ W, unsigned short* __restrict__ WT,
                                 int K, int Ncols) {
    int idx = blockIdx.x * 256 + threadIdx.x;
    if (idx < K * Ncols) {
        int k = idx / Ncols, n = idx - k * Ncols;
        WT[n * K + k] = f2bf(W[idx]);
    }
}

__global__ void k_zero_f32(float* __restrict__ p, long long n) {
    long long i = (long long)blockIdx.x * 256 + threadIdx.x;
    if (i < n) p[i] = 0.0f;
}

// ---------------------------------------------------------------- WMMA GEMM
// C[M][Ncols] = A[M][K](bf16,row-major) * B (supplied as Bt[Ncols][K], bf16)
// One wave computes a 16x64 output tile (4 N-subtiles) so the A fragment is
// reused 4x per K-step: 10 b128 loads per 4 v_wmma_f32_16x16x32_bf16.
__global__ void k_gemm_bf16(const unsigned short* __restrict__ A,
                            const unsigned short* __restrict__ Bt,
                            float* __restrict__ C,
                            int M, int Ncols, int K) {
    const int wave   = threadIdx.x >> 5;
    const int lane   = threadIdx.x & 31;
    const int tilesN = Ncols >> 6;                       // 64-wide N tiles
    const int tile   = blockIdx.x * (blockDim.x >> 5) + wave;
    const int mt = tile / tilesN;
    const int nt = tile - mt * tilesN;
    if (mt * 16 >= M) return;

    const int lh   = lane & 15;
    const int aoff = (lane < 16) ? 0 : 16;   // A: lanes 16-31 hold K+16..31
    const int boff = (lane < 16) ? 0 : 16;   // B: lanes 16-31 hold K+16..31
    const unsigned short* arow = A + (size_t)(mt * 16 + lh) * K;
    const unsigned short* bc0 = Bt + (size_t)(nt * 64 + 0  + lh) * K + boff;
    const unsigned short* bc1 = Bt + (size_t)(nt * 64 + 16 + lh) * K + boff;
    const unsigned short* bc2 = Bt + (size_t)(nt * 64 + 32 + lh) * K + boff;
    const unsigned short* bc3 = Bt + (size_t)(nt * 64 + 48 + lh) * K + boff;

    v8f acc0 = {}, acc1 = {}, acc2 = {}, acc3 = {};
    for (int k = 0; k < K; k += 32) {
        u16x8 alo = *(const u16x8*)(arow + k + aoff);
        u16x8 ahi = *(const u16x8*)(arow + k + 8 + aoff);
        BF16Frag a, b0, b1, b2, b3;
        a.u  = __builtin_shufflevector(alo, ahi, 0,1,2,3,4,5,6,7,8,9,10,11,12,13,14,15);
        b0.u = *(const u16x16*)(bc0 + k);
        b1.u = *(const u16x16*)(bc1 + k);
        b2.u = *(const u16x16*)(bc2 + k);
        b3.u = *(const u16x16*)(bc3 + k);
        acc0 = __builtin_amdgcn_wmma_f32_16x16x32_bf16(false, a.v, false, b0.v, (short)0, acc0, false, false);
        acc1 = __builtin_amdgcn_wmma_f32_16x16x32_bf16(false, a.v, false, b1.v, (short)0, acc1, false, false);
        acc2 = __builtin_amdgcn_wmma_f32_16x16x32_bf16(false, a.v, false, b2.v, (short)0, acc2, false, false);
        acc3 = __builtin_amdgcn_wmma_f32_16x16x32_bf16(false, a.v, false, b3.v, (short)0, acc3, false, false);
    }
    // D layout: VGPR r -> row r (lanes 0-15) / row 8+r (lanes 16-31), col = lane&15
    float* crow = C + (size_t)(mt * 16 + ((lane < 16) ? 0 : 8)) * Ncols + nt * 64 + lh;
#pragma unroll
    for (int r = 0; r < 8; ++r) {
        crow[(size_t)r * Ncols + 0]  = acc0[r];
        crow[(size_t)r * Ncols + 16] = acc1[r];
        crow[(size_t)r * Ncols + 32] = acc2[r];
        crow[(size_t)r * Ncols + 48] = acc3[r];
    }
}

// out[128][128] += X2T_tile_m ^ . * X2T_tile_n  over a K chunk (split-K, f32 atomics)
__global__ void k_gemm_xtx(const unsigned short* __restrict__ X2T,
                           float* __restrict__ out, int kChunk) {
    const int lane = threadIdx.x & 31;
    const int mt = blockIdx.x >> 3;
    const int nt = blockIdx.x & 7;
    int k0 = blockIdx.y * kChunk;
    int k1 = k0 + kChunk; if (k1 > KPAD) k1 = KPAD;
    if (k0 >= k1) return;

    const int lh   = lane & 15;
    const int aoff = (lane < 16) ? 0 : 16;
    const int boff = (lane < 16) ? 0 : 16;
    const unsigned short* arow = X2T + (size_t)(mt * 16 + lh) * KPAD;
    const unsigned short* bcol = X2T + (size_t)(nt * 16 + lh) * KPAD;

    v8f acc = {};
    for (int k = k0; k < k1; k += 32) {
        u16x8 alo = *(const u16x8*)(arow + k + aoff);
        u16x8 ahi = *(const u16x8*)(arow + k + 8 + aoff);
        BF16Frag a, b;
        a.u = __builtin_shufflevector(alo, ahi, 0,1,2,3,4,5,6,7,8,9,10,11,12,13,14,15);
        b.u = *(const u16x16*)(bcol + k + boff);
        acc = __builtin_amdgcn_wmma_f32_16x16x32_bf16(false, a.v, false, b.v,
                                                      (short)0, acc, false, false);
    }
    float* crow = out + (size_t)(mt * 16 + ((lane < 16) ? 0 : 8)) * F_OUT + nt * 16 + lh;
#pragma unroll
    for (int r = 0; r < 8; ++r) atomicAdd(&crow[(size_t)r * F_OUT], acc[r]);
}

// ---------------------------------------------------------------- edge scatter
// AGG[dst] += H[src] * coef[e]; thread = (edge, 4-feature group)
__global__ void k_edge_agg(const long long* __restrict__ ei, const float* __restrict__ coef,
                           const float* __restrict__ H, float* __restrict__ AGG,
                           int F, int shift) {
    long long tid = (long long)blockIdx.x * blockDim.x + threadIdx.x;
    long long e = tid >> shift;
    if (e >= N_EDGES) return;
    int f4 = (int)(tid & ((1 << shift) - 1)) << 2;
    int s = (int)ei[e];
    int d = (int)ei[N_EDGES + e];
    float c = coef[e];
    const float4 h = *(const float4*)(H + (size_t)s * F + f4);
    float* ap = AGG + (size_t)d * F + f4;
    atomicAdd(ap + 0, h.x * c);
    atomicAdd(ap + 1, h.y * c);
    atomicAdd(ap + 2, h.z * c);
    atomicAdd(ap + 3, h.w * c);
}

// ---------------------------------------------------------------- fused epilogues
// x1 = sigmoid(BN(relu(agg + dinv^2*h + b))) -> bf16 row-major [N][256]
__global__ void k_ew1(const float* __restrict__ AGG, const float* __restrict__ H,
                      const float* __restrict__ dinv, const float* __restrict__ b,
                      const float* __restrict__ g,  const float* __restrict__ be,
                      const float* __restrict__ m,  const float* __restrict__ v,
                      unsigned short* __restrict__ X1B) {
    long long i = (long long)blockIdx.x * 256 + threadIdx.x;
    if (i >= (long long)N_NODES * F_HID) return;
    int node = (int)(i >> 8);
    int f    = (int)(i & (F_HID - 1));
    float di = dinv[node];
    float val = AGG[i] + di * di * H[i] + b[f];
    val = fmaxf(val, 0.0f);
    val = (val - m[f]) * rsqrtf(v[f] + BN_EPS) * g[f] + be[f];
    val = 1.0f / (1.0f + __expf(-val));
    X1B[i] = f2bf(val);
}

// x2 epilogue, written TRANSPOSED: X2T[f][node] bf16, stride KPAD
__global__ void k_ew2(const float* __restrict__ AGG, const float* __restrict__ H,
                      const float* __restrict__ dinv, const float* __restrict__ b,
                      const float* __restrict__ g,  const float* __restrict__ be,
                      const float* __restrict__ m,  const float* __restrict__ v,
                      unsigned short* __restrict__ X2T) {
    long long i = (long long)blockIdx.x * 256 + threadIdx.x;
    if (i >= (long long)N_NODES * F_OUT) return;
    int node = (int)(i >> 7);
    int f    = (int)(i & (F_OUT - 1));
    float di = dinv[node];
    float val = AGG[i] + di * di * H[i] + b[f];
    val = fmaxf(val, 0.0f);
    val = (val - m[f]) * rsqrtf(v[f] + BN_EPS) * g[f] + be[f];
    val = 1.0f / (1.0f + __expf(-val));
    X2T[(size_t)f * KPAD + node] = f2bf(val);
}

__global__ void k_pad_x2t(unsigned short* __restrict__ X2T) {
    int i = blockIdx.x * 256 + threadIdx.x;    // 128*16 tail elements
    if (i < F_OUT * (KPAD - N_NODES)) {
        int f = i >> 4, j = i & 15;
        X2T[(size_t)f * KPAD + N_NODES + j] = 0;
    }
}

// ---------------------------------------------------------------- launch
extern "C" void kernel_launch(void* const* d_in, const int* in_sizes, int n_in,
                              void* d_out, int out_size, void* d_ws, size_t ws_size,
                              hipStream_t stream) {
    const float*     x   = (const float*)d_in[0];
    const long long* ei  = (const long long*)d_in[1];   // int64 [2][E]
    const float*     W1  = (const float*)d_in[2];
    const float*     b1  = (const float*)d_in[3];
    const float*     W2  = (const float*)d_in[4];
    const float*     b2  = (const float*)d_in[5];
    /* d_in[6]=W3, d_in[7]=b3: dead in reference forward */
    const float*     g1  = (const float*)d_in[8];
    const float*     be1 = (const float*)d_in[9];
    const float*     m1  = (const float*)d_in[10];
    const float*     v1  = (const float*)d_in[11];
    const float*     g2  = (const float*)d_in[12];
    const float*     be2 = (const float*)d_in[13];
    const float*     m2  = (const float*)d_in[14];
    const float*     v2  = (const float*)d_in[15];
    float* out = (float*)d_out;

    // workspace carve-up (256B aligned), with buffer reuse across phases
    char*  w = (char*)d_ws;
    size_t o = 0;
    auto carve = [&](size_t bytes) { void* p = w + o; o += (bytes + 255) & ~(size_t)255; return p; };
    float*          dinv = (float*)carve((size_t)N_NODES * 4);
    float*          coef = (float*)carve((size_t)N_EDGES * 4);
    unsigned short* W1T  = (unsigned short*)carve((size_t)F_IN * F_HID * 2);
    unsigned short* W2T  = (unsigned short*)carve((size_t)F_HID * F_OUT * 2);
    float*          h    = (float*)carve((size_t)N_NODES * F_HID * 4);  // h1, then h2
    float*          agg  = (float*)carve((size_t)N_NODES * F_HID * 4);  // agg1, then agg2
    unsigned short* x1b  = (unsigned short*)carve((size_t)N_NODES * F_HID * 2);
    unsigned short* xb   = (unsigned short*)carve((size_t)F_OUT * KPAD * 2); // xb, then x2T
    unsigned short* x2t  = xb;

    const int T = 256;
    // degree / dinv / per-edge coefficient
    k_deg_init <<<(N_NODES + T - 1) / T, T, 0, stream>>>(dinv);
    k_deg_count<<<(N_EDGES + T - 1) / T, T, 0, stream>>>(ei, dinv);
    k_dinv     <<<(N_NODES + T - 1) / T, T, 0, stream>>>(dinv);
    k_edge_coef<<<(N_EDGES + T - 1) / T, T, 0, stream>>>(ei, dinv, coef);

    // precision conversion / weight transpose (B column-major for WMMA)
    k_f32_to_bf16  <<<(int)(((long long)N_NODES * F_IN + T - 1) / T), T, 0, stream>>>(x, xb, (long long)N_NODES * F_IN);
    k_transpose_bf16<<<(F_IN * F_HID + T - 1) / T, T, 0, stream>>>(W1, W1T, F_IN, F_HID);
    k_transpose_bf16<<<(F_HID * F_OUT + T - 1) / T, T, 0, stream>>>(W2, W2T, F_HID, F_OUT);

    // layer 1: GEMM -> scatter -> fused epilogue
    {
        int tiles = (N_NODES / 16) * (F_HID / 64);           // 16x64 tiles per wave
        k_gemm_bf16<<<(tiles + 7) / 8, T, 0, stream>>>(xb, W1T, h, N_NODES, F_HID, F_IN);
    }
    k_zero_f32 <<<(int)(((long long)N_NODES * F_HID + T - 1) / T), T, 0, stream>>>(agg, (long long)N_NODES * F_HID);
    k_edge_agg <<<(int)(((long long)N_EDGES * (F_HID / 4) + T - 1) / T), T, 0, stream>>>(ei, coef, h, agg, F_HID, 6);
    k_ew1      <<<(int)(((long long)N_NODES * F_HID + T - 1) / T), T, 0, stream>>>(agg, h, dinv, b1, g1, be1, m1, v1, x1b);

    // layer 2: GEMM -> scatter -> fused epilogue (writes x2 transposed, bf16)
    {
        int tiles = (N_NODES / 16) * (F_OUT / 64);
        k_gemm_bf16<<<(tiles + 7) / 8, T, 0, stream>>>(x1b, W2T, h, N_NODES, F_OUT, F_HID);
    }
    k_zero_f32 <<<(int)(((long long)N_NODES * F_OUT + T - 1) / T), T, 0, stream>>>(agg, (long long)N_NODES * F_OUT);
    k_edge_agg <<<(int)(((long long)N_EDGES * (F_OUT / 4) + T - 1) / T), T, 0, stream>>>(ei, coef, h, agg, F_OUT, 5);
    k_ew2      <<<(int)(((long long)N_NODES * F_OUT + T - 1) / T), T, 0, stream>>>(agg, h, dinv, b2, g2, be2, m2, v2, x2t);
    k_pad_x2t  <<<(F_OUT * (KPAD - N_NODES) + T - 1) / T, T, 0, stream>>>(x2t);

    // x3 = x2^T x2 : split-K WMMA with f32 atomic reduction into d_out
    k_zero_f32 <<<(F_OUT * F_OUT + T - 1) / T, T, 0, stream>>>(out, (long long)F_OUT * F_OUT);
    k_gemm_xtx <<<dim3((F_OUT / 16) * (F_OUT / 16), 32), 32, 0, stream>>>(x2t, out, 1568);
}
